// TXCDRBlockSparseTopK_90984587198480
// MI455X (gfx1250) — compile-verified, matrix operand order checked by reading, not run on hardware
//
#include <hip/hip_runtime.h>

// ---------------------------------------------------------------------------
// Problem constants (match reference)
// ---------------------------------------------------------------------------
#define BB    256
#define TT    8
#define DIN   768
#define DSAE  16384
#define KTOP  64
#define NFLAT (TT * DSAE)        // 131072 flattened features per batch row

// Encoder tiling: one workgroup computes full M=256 (batch) x N=64 (sae cols)
// for one t, so W_enc is streamed from HBM exactly once (402 MB -> ~17us at
// 23.3 TB/s).  K is stepped in 32 (bf16 WMMA depth).
#define NTILE 64
#define KTILE 32
#define LSTR  20                 // bf16 elements per lane in LDS fragment (16 + 4 pad)
#define GSTR  (32 * LSTR)        // elements per 16x32 fragment group in LDS

typedef __attribute__((ext_vector_type(16))) __bf16    v16bf;
typedef __attribute__((ext_vector_type(8)))  float     v8f;
typedef __attribute__((ext_vector_type(2)))  unsigned  uint2v;
typedef __attribute__((ext_vector_type(4)))  unsigned  u32x4;
typedef __attribute__((ext_vector_type(8)))  int       i32x8;
typedef __attribute__((ext_vector_type(4)))  int       i32x4;

union FragU { uint2v u[4]; v16bf v; };

// TDM (tensor data mover) path for the sparse-decode gather.  This toolchain
// exposes the 6-arg builtin (clang-23/therock lane):
//   (u32x4 g0, i32x8 g1, i32x4 g2, i32x4 g3, i32x8, i32 cpol)
#if __has_builtin(__builtin_amdgcn_tensor_load_to_lds) && \
    __has_builtin(__builtin_amdgcn_s_wait_tensorcnt)
#define HAVE_TDM 1
#endif

// key transform: monotonic float ordering in unsigned space
__device__ __forceinline__ unsigned fkey(float f) {
    unsigned u = __float_as_uint(f);
    return (u & 0x80000000u) ? ~u : (u | 0x80000000u);
}

// ---------------------------------------------------------------------------
// Kernel 1: encoder GEMM  pre[b,t,s] = sum_k x[b,t,k] * W_enc[t,k,s] + b_enc[s]
// bf16 WMMA 16x16x32, fp32 accumulate.  Output written into the z-region of
// d_out (reused as scratch for pre; later zeroed + scattered).
// ---------------------------------------------------------------------------
__global__ __launch_bounds__(256) void encoder_kernel(
    const float* __restrict__ x, const float* __restrict__ W_enc,
    const float* __restrict__ b_enc, float* __restrict__ pre)
{
    // A: 16 groups (16 rows each) of 16x32 bf16 fragments, fragment-major
    __shared__ __align__(16) __bf16 Asm[16 * GSTR];   // 20.0 KB
    // B: 4 groups (16 cols each) of 32x16 bf16 fragments, fragment-major
    __shared__ __align__(16) __bf16 Bsm[4 * GSTR];    //  5.0 KB

    const int t    = blockIdx.x / (DSAE / NTILE);
    const int n0   = (blockIdx.x % (DSAE / NTILE)) * NTILE;
    const int tid  = threadIdx.x;
    const int wave = tid >> 5;
    const int lane = tid & 31;
    const int l16  = lane & 15;

    v8f acc[2][4];
#pragma unroll
    for (int mi = 0; mi < 2; ++mi)
#pragma unroll
        for (int nj = 0; nj < 4; ++nj)
#pragma unroll
            for (int r = 0; r < 8; ++r) acc[mi][nj][r] = 0.f;

    for (int k0 = 0; k0 < DIN; k0 += KTILE) {
        // ---- load & convert A tile: 256 rows x 32 k of x (f32 -> bf16) ----
        {
            const int row = tid;
            const float4* src =
                (const float4*)(x + ((size_t)row * TT + t) * DIN + k0);
            const int g = row >> 4, l = row & 15;
#pragma unroll
            for (int i = 0; i < 8; ++i) {
                float4 v = src[i];
                const int kb = i * 4;
#pragma unroll
                for (int j = 0; j < 4; ++j) {
                    const int k  = kb + j;
                    const int lh = (k >> 3) & 1;
                    const int e  = (k & 7) | (((k >> 4) & 1) << 3);
                    float f = (j == 0) ? v.x : (j == 1) ? v.y : (j == 2) ? v.z : v.w;
                    Asm[g * GSTR + (lh * 16 + l) * LSTR + e] = (__bf16)f;
                }
            }
        }
        // ---- load & convert B tile: 32 k x 64 n of W_enc (f32 -> bf16) ----
        {
#pragma unroll
            for (int i = 0; i < 8; ++i) {
                const int e2 = tid + i * 256;
                const int k  = e2 >> 6;
                const int nn = e2 & 63;
                float w = W_enc[((size_t)t * DIN + k0 + k) * DSAE + n0 + nn];
                const int nj = nn >> 4, l = nn & 15;
                const int lh = (k >> 3) & 1;
                const int e  = (k & 7) | (((k >> 4) & 1) << 3);
                Bsm[nj * GSTR + (lh * 16 + l) * LSTR + e] = (__bf16)w;
            }
        }
        __syncthreads();

        // ---- fragments + 8 WMMA per wave ----
        FragU a[2], bf[4];
#pragma unroll
        for (int mi = 0; mi < 2; ++mi) {
            const int g = wave * 2 + mi;
            const uint2v* ap = (const uint2v*)(Asm + g * GSTR + lane * LSTR);
#pragma unroll
            for (int q = 0; q < 4; ++q) a[mi].u[q] = ap[q];
        }
#pragma unroll
        for (int nj = 0; nj < 4; ++nj) {
            const uint2v* bp = (const uint2v*)(Bsm + nj * GSTR + lane * LSTR);
#pragma unroll
            for (int q = 0; q < 4; ++q) bf[nj].u[q] = bp[q];
        }
#pragma unroll
        for (int mi = 0; mi < 2; ++mi)
#pragma unroll
            for (int nj = 0; nj < 4; ++nj)
                acc[mi][nj] = __builtin_amdgcn_wmma_f32_16x16x32_bf16(
                    false, a[mi].v, false, bf[nj].v, (short)0, acc[mi][nj],
                    false, false);
        __syncthreads();
    }

    // ---- store: C layout = VGPR r -> M=(lane>=16?8:0)+r, N=lane%16 ----
    const int lh = lane >> 4;
#pragma unroll
    for (int nj = 0; nj < 4; ++nj) {
        const int colg = n0 + nj * 16 + l16;
        const float be = b_enc[colg];
#pragma unroll
        for (int mi = 0; mi < 2; ++mi) {
#pragma unroll
            for (int r = 0; r < 8; ++r) {
                const int rowg = wave * 32 + mi * 16 + lh * 8 + r;
                pre[((size_t)rowg * TT + t) * DSAE + colg] = acc[mi][nj][r] + be;
            }
        }
    }
}

// ---------------------------------------------------------------------------
// Kernel 2: per-batch-row top-64 via 4-bit radix select (8 rounds over the
// 131072-entry slice, L2-resident).  Deterministic lowest-index tie-breaking
// via ordered per-segment counts + serial exclusive scan.
// ---------------------------------------------------------------------------
__global__ __launch_bounds__(256) void topk_kernel(
    const float* __restrict__ pre, int* __restrict__ tk_idx,
    float* __restrict__ tk_val)
{
    const int b   = blockIdx.x;
    const int tid = threadIdx.x;
    const float* p = pre + (size_t)b * NFLAT;

    __shared__ unsigned cnt[16];
    __shared__ int gtb[256];
    __shared__ int tieb[256];
    __shared__ int tot_gt_sh;

    unsigned prefix = 0;
    int remaining = KTOP;
    for (int shift = 28; shift >= 0; shift -= 4) {
        if (tid < 16) cnt[tid] = 0u;
        __syncthreads();
        const unsigned himask = (shift == 28) ? 0u : (0xFFFFFFFFu << (shift + 4));
        for (int i = tid; i < NFLAT; i += 256) {
            const unsigned key = fkey(p[i]);
            if ((key & himask) == (prefix & himask))
                atomicAdd(&cnt[(key >> shift) & 15u], 1u);
        }
        __syncthreads();
        unsigned chi = 0; int d = 0;
        for (int dd = 15; dd >= 0; --dd) {
            if (chi + cnt[dd] >= (unsigned)remaining) { d = dd; break; }
            chi += cnt[dd];
        }
        prefix |= ((unsigned)d) << shift;
        remaining -= (int)chi;
        __syncthreads();
    }
    const unsigned Tk = prefix;           // exact key of the k-th largest

    // ordered selection: thread tid owns contiguous segment [tid*512, +512)
    const int seg  = NFLAT / 256;
    const int base = tid * seg;
    int cg = 0, ct = 0;
    for (int i = 0; i < seg; ++i) {
        const unsigned key = fkey(p[base + i]);
        cg += (key > Tk);
        ct += (key == Tk);
    }
    gtb[tid] = cg; tieb[tid] = ct;
    __syncthreads();
    if (tid == 0) {                       // serial exclusive scan (deterministic)
        int rg = 0, rt = 0;
        for (int i = 0; i < 256; ++i) {
            int g = gtb[i];  gtb[i]  = rg; rg += g;
            int s = tieb[i]; tieb[i] = rt; rt += s;
        }
        tot_gt_sh = rg;
    }
    __syncthreads();
    int posg = gtb[tid];
    int post = tieb[tid];
    const int tot_gt = tot_gt_sh;
    const int need = KTOP - tot_gt;       // ties to keep, lowest index first
    for (int i = 0; i < seg; ++i) {
        const float v = p[base + i];
        const unsigned key = fkey(v);
        if (key > Tk) {
            tk_idx[b * KTOP + posg] = base + i;
            tk_val[b * KTOP + posg] = fmaxf(v, 0.f);
            ++posg;
        } else if (key == Tk) {
            if (post < need) {
                tk_idx[b * KTOP + tot_gt + post] = base + i;
                tk_val[b * KTOP + tot_gt + post] = fmaxf(v, 0.f);
            }
            ++post;
        }
    }
}

// ---------------------------------------------------------------------------
// Kernel 3: zero the z region (it currently holds pre).  Region start is only
// 4B-aligned inside d_out, so peel 3 floats then float4 stores.
// ---------------------------------------------------------------------------
__global__ __launch_bounds__(256) void zero_kernel(float* __restrict__ z)
{
    const size_t ZN = (size_t)BB * NFLAT;
    const size_t N4 = (ZN - 3) / 4;
    const size_t i  = (size_t)blockIdx.x * blockDim.x + threadIdx.x;
    float4* z4 = (float4*)(z + 3);
    if (i < N4) z4[i] = make_float4(0.f, 0.f, 0.f, 0.f);
    if (i == 0) { z[0] = z[1] = z[2] = 0.f; z[ZN - 1] = 0.f; }
}

// ---------------------------------------------------------------------------
// Kernel 4: scatter relu(top-k values) into z
// ---------------------------------------------------------------------------
__global__ __launch_bounds__(256) void scatter_kernel(
    const int* __restrict__ tk_idx, const float* __restrict__ tk_val,
    float* __restrict__ z)
{
    const int g = blockIdx.x * blockDim.x + threadIdx.x;   // 0 .. B*KTOP-1
    const int b = g / KTOP;
    z[(size_t)b * NFLAT + tk_idx[g]] = tk_val[g];
}

// ---------------------------------------------------------------------------
// Kernel 5: sparse decode + per-batch squared error.
// One block per b: acc[T*DIN] in LDS.  The 64 gathered W_dec rows (3 KB f32
// each, no conversion needed) are staged LDS-side by the Tensor Data Mover:
// wave 0 issues a 1-D tensor_load_to_lds descriptor for row e+1 while all
// waves FMA row e from the other LDS buffer; completion via s_wait_tensorcnt
// (TDM ops from one wave complete in order).  Fallback: direct loads+prefetch.
// ---------------------------------------------------------------------------
__global__ __launch_bounds__(256) void decode_kernel(
    const float* __restrict__ x, const float* __restrict__ W_dec,
    const float* __restrict__ b_dec, const int* __restrict__ tk_idx,
    const float* __restrict__ tk_val, float* __restrict__ x_hat,
    float* __restrict__ loss_partial)
{
    __shared__ float acc[TT * DIN];       // 24 KB
    __shared__ float red[256];
#if defined(HAVE_TDM)
    __shared__ __align__(16) float wbuf[2][DIN];   // 2 x 3 KB TDM staging
#endif
    const int b = blockIdx.x, tid = threadIdx.x;

    for (int i = tid; i < TT * DIN; i += 256) acc[i] = b_dec[i];

#if defined(HAVE_TDM)
    // scalar (wave-uniform) branch: TDM issues regardless of EXEC, so gate
    // with an SGPR condition -> s_cbranch skips it in waves 1..7 entirely.
    const bool wave0 = (__builtin_amdgcn_readfirstlane((int)tid) < 32);

    // Issue a 1-D TDM copy: 768 x 4B row of W_dec -> LDS buffer.
    auto tdm_issue_row = [&](int e, int buf) {
        const int   idx = tk_idx[b * KTOP + e];
        const int   t = idx / DSAE, s = idx % DSAE;
        const unsigned long long ga =
            (unsigned long long)(const void*)(W_dec + ((size_t)s * TT + t) * DIN);
        const unsigned lds_off =
            (unsigned)(unsigned long long)(const void*)&wbuf[buf][0];
        const unsigned ga_lo =
            (unsigned)__builtin_amdgcn_readfirstlane((int)(unsigned)ga);
        const unsigned ga_hi =
            (unsigned)__builtin_amdgcn_readfirstlane((int)(unsigned)(ga >> 32));
        const unsigned lo32 =
            (unsigned)__builtin_amdgcn_readfirstlane((int)lds_off);
        // D# group 0: count=1 | lds_addr[63:32] | global_addr[120:64] | type=2
        u32x4 g0;
        g0[0] = 1u;                                  // count=1, user descriptor
        g0[1] = lo32;                                // LDS byte address
        g0[2] = ga_lo;                               // global addr [31:0]
        g0[3] = (ga_hi & 0x01FFFFFFu) | 0x80000000u; // addr[56:32] | type=2
        // D# group 1: data_size=4B; tensor_dim0=tile_dim0=768; stride0=768;
        // tensor_dim1=1; 1-D tile (tile_dim1=tile_dim2=0); no pad/iterate.
        i32x8 g1;
        g1[0] = (int)(2u << 16);                     // data_size = 4 bytes
        g1[1] = (int)((DIN & 0xFFFFu) << 16);        // tensor_dim0[15:0]
        g1[2] = (int)((DIN >> 16) | (1u << 16));     // td0[31:16] | tensor_dim1
        g1[3] = (int)((DIN & 0xFFFFu) << 16);        // td1[31:16]=0 | tile_dim0
        g1[4] = 0;                                   // tile_dim1=0, tile_dim2=0
        g1[5] = DIN;                                 // tensor_dim0_stride
        g1[6] = 0;
        g1[7] = 0;
        const i32x4 gz4 = {0, 0, 0, 0};              // groups 2/3 unused (1-D)
        const i32x8 gz8 = {0, 0, 0, 0, 0, 0, 0, 0};  // extra group (6-arg form)
        __builtin_amdgcn_tensor_load_to_lds(g0, g1, gz4, gz4, gz8, 0);
    };

    if (wave0) tdm_issue_row(0, 0);                  // prologue: stage row 0
    __syncthreads();                                 // acc init published

    for (int e = 0; e < KTOP; ++e) {
        const int buf = e & 1;
        if (wave0) {
            if (e + 1 < KTOP) {
                tdm_issue_row(e + 1, buf ^ 1);       // overlap next gather
                __builtin_amdgcn_s_wait_tensorcnt((short)1);  // row e done
            } else {
                __builtin_amdgcn_s_wait_tensorcnt((short)0);
            }
        }
        __syncthreads();                             // publish wbuf[buf]

        const int   idx = tk_idx[b * KTOP + e];
        const float v   = tk_val[b * KTOP + e];
        const int t = idx / DSAE;
#pragma unroll
        for (int u = 0; u < 3; ++u) {                // thread tid owns tid+256u
            const int j = tid + u * 256;
            acc[t * DIN + j] = fmaf(v, wbuf[buf][j], acc[t * DIN + j]);
        }
        __syncthreads();                             // done reading wbuf[buf]
    }
#else
    __syncthreads();
    for (int e = 0; e < KTOP; ++e) {
        const int   idx = tk_idx[b * KTOP + e];
        const float v   = tk_val[b * KTOP + e];
        const int t = idx / DSAE, s = idx % DSAE;
        const float* wr = W_dec + ((size_t)s * TT + t) * DIN;
        if (e + 1 < KTOP) {               // prefetch next gathered row
            const int idx2 = tk_idx[b * KTOP + e + 1];
            const int t2 = idx2 / DSAE, s2 = idx2 % DSAE;
            __builtin_prefetch(W_dec + ((size_t)s2 * TT + t2) * DIN + tid * 3, 0, 0);
        }
#pragma unroll
        for (int u = 0; u < 3; ++u) {     // thread tid owns offsets tid+256u
            const int j = tid + u * 256;
            acc[t * DIN + j] = fmaf(v, wr[j], acc[t * DIN + j]);
        }
    }
    __syncthreads();
#endif

    float err = 0.f;
    const float* xb = x + (size_t)b * TT * DIN;
    float*       xh = x_hat + (size_t)b * TT * DIN;
    for (int i = tid; i < TT * DIN; i += 256) {
        const float h = acc[i];
        xh[i] = h;
        const float d = h - xb[i];
        err = fmaf(d, d, err);
    }
    red[tid] = err;
    __syncthreads();
    if (tid == 0) {                       // fixed-order reduce (deterministic)
        float s = 0.f;
        for (int i = 0; i < 256; ++i) s += red[i];
        loss_partial[b] = s;
    }
}

// ---------------------------------------------------------------------------
// Kernel 6: final loss = sum(loss_partial) / (B*T), fixed order
// ---------------------------------------------------------------------------
__global__ void loss_kernel(const float* __restrict__ lp, float* __restrict__ out)
{
    if (blockIdx.x == 0 && threadIdx.x == 0) {
        float s = 0.f;
        for (int i = 0; i < BB; ++i) s += lp[i];
        out[0] = s / (float)(BB * TT);
    }
}

// ---------------------------------------------------------------------------
extern "C" void kernel_launch(void* const* d_in, const int* in_sizes, int n_in,
                              void* d_out, int out_size, void* d_ws, size_t ws_size,
                              hipStream_t stream)
{
    const float* x     = (const float*)d_in[0];   // (B,T,DIN)
    const float* W_enc = (const float*)d_in[1];   // (T,DIN,DSAE)
    const float* W_dec = (const float*)d_in[2];   // (DSAE,T,DIN)
    const float* b_enc = (const float*)d_in[3];   // (DSAE)
    const float* b_dec = (const float*)d_in[4];   // (T,DIN)
    (void)in_sizes; (void)n_in; (void)out_size; (void)ws_size;

    float* out   = (float*)d_out;
    float* loss  = out;                               // [1]
    float* x_hat = out + 1;                           // [B*T*DIN]
    float* z     = out + 1 + (size_t)BB * TT * DIN;   // [B*T*DSAE]; also 'pre'

    int*   tk_idx = (int*)d_ws;                       // B*KTOP ints
    float* tk_val = (float*)d_ws + (size_t)BB * KTOP; // B*KTOP floats
    float* lp     = tk_val + (size_t)BB * KTOP;       // B floats

    // 1) encoder: pre -> z region (WMMA bf16, W_enc streamed once)
    encoder_kernel<<<dim3(TT * (DSAE / NTILE)), dim3(256), 0, stream>>>(
        x, W_enc, b_enc, z);
    // 2) top-64 per batch row (radix select, deterministic ties)
    topk_kernel<<<dim3(BB), dim3(256), 0, stream>>>(z, tk_idx, tk_val);
    // 3) zero z, 4) scatter relu(topk) back in
    {
        const size_t N4 = ((size_t)BB * NFLAT - 3) / 4;
        zero_kernel<<<dim3((unsigned)((N4 + 255) / 256)), dim3(256), 0, stream>>>(z);
    }
    scatter_kernel<<<dim3(BB * KTOP / 256), dim3(256), 0, stream>>>(tk_idx, tk_val, z);
    // 5) sparse decode (TDM-gathered W_dec rows) + per-b loss, 6) final loss
    decode_kernel<<<dim3(BB), dim3(256), 0, stream>>>(
        x, W_dec, b_dec, tk_idx, tk_val, x_hat, lp);
    loss_kernel<<<dim3(1), dim3(1), 0, stream>>>(lp, loss);
}
